// MultiHeadAttention_42614665511259
// MI455X (gfx1250) — compile-verified, matrix operand order checked by reading, not run on hardware
//
#include <hip/hip_runtime.h>
#include <hip/hip_bf16.h>

typedef __bf16 bf16_t;
typedef __attribute__((ext_vector_type(16))) __bf16 v16bf;
typedef __attribute__((ext_vector_type(8)))  __bf16 v8bf;
typedef __attribute__((ext_vector_type(8)))  float  v8f;
typedef __attribute__((ext_vector_type(4)))  float  v4f;

#define SEQ    2048
#define DMODEL 1024
#define NH     16
#define HD     64

union V16U { v16bf v; v8bf h[2]; };

// ---------------------------------------------------------------------------
// Kernel 1: projection GEMM  (X[2048x1024] * W[1024x1024] + b), one instance
// per output matrix. TRANSPOSED=false -> out[m][n] bf16 (Q, K);
// TRANSPOSED=true -> out[n][m] bf16 (V stored transposed for the AV GEMM).
// Block tile 128(M) x 64(N), k-step 32, 8 waves each owning a 32x32 (2x2 WMMA).
// ---------------------------------------------------------------------------
template <bool TRANSPOSED>
__global__ __launch_bounds__(256) void proj_gemm_kernel(
    const float* __restrict__ x,
    const float* __restrict__ W, const float* __restrict__ bia,
    bf16_t* __restrict__ outp)
{
    __shared__ bf16_t As[128][40];   // [m][k], 80B row stride keeps 16B align
    __shared__ bf16_t Bs[64][40];    // [n][k] (W tile transposed)

    const int t    = threadIdx.x;
    const int lane = t & 31;
    const int wave = t >> 5;
    const int half = lane >> 4;
    const int ln   = lane & 15;

    const int m0 = blockIdx.y * 128;
    const int n0 = blockIdx.x * 64;

    const int wm = (wave >> 1) * 32;   // wave row within block tile
    const int wn = (wave & 1) * 32;    // wave col within block tile

    v8f c[2][2];
    c[0][0] = {}; c[0][1] = {}; c[1][0] = {}; c[1][1] = {};

    const int arow = t >> 1;           // 0..127
    const int acol = (t & 1) * 16;     // 0 or 16
    const int brow = t >> 3;           // 0..31 (k within tile)
    const int bcol = (t & 7) * 8;      // n segment

    for (int kb = 0; kb < DMODEL; kb += 32) {
        // stage A tile (fp32 -> bf16)
        {
            const float* src = x + (size_t)(m0 + arow) * DMODEL + kb + acol;
            __builtin_prefetch(src + 32, 0, 0);   // next k-tile of X
            v4f f0 = *reinterpret_cast<const v4f*>(src);
            v4f f1 = *reinterpret_cast<const v4f*>(src + 4);
            v4f f2 = *reinterpret_cast<const v4f*>(src + 8);
            v4f f3 = *reinterpret_cast<const v4f*>(src + 12);
            v8bf lo, hi;
            #pragma unroll
            for (int i = 0; i < 4; ++i) {
                lo[i]   = (bf16_t)f0[i]; lo[i+4] = (bf16_t)f1[i];
                hi[i]   = (bf16_t)f2[i]; hi[i+4] = (bf16_t)f3[i];
            }
            v8bf* dst = reinterpret_cast<v8bf*>(&As[arow][acol]);
            dst[0] = lo; dst[1] = hi;
        }
        // stage B tile transposed: Bs[n][k] = W[kb+k][n0+n]
        {
            const float* src = W + (size_t)(kb + brow) * DMODEL + n0 + bcol;
            __builtin_prefetch(src + 32 * DMODEL, 0, 0);   // next k-tile of W
            v4f f0 = *reinterpret_cast<const v4f*>(src);
            v4f f1 = *reinterpret_cast<const v4f*>(src + 4);
            #pragma unroll
            for (int i = 0; i < 4; ++i) {
                Bs[bcol + i][brow]     = (bf16_t)f0[i];
                Bs[bcol + 4 + i][brow] = (bf16_t)f1[i];
            }
        }
        __syncthreads();

        V16U a0, a1, b0, b1;
        a0.h[0] = *reinterpret_cast<const v8bf*>(&As[wm + ln][8 * half]);
        a0.h[1] = *reinterpret_cast<const v8bf*>(&As[wm + ln][16 + 8 * half]);
        a1.h[0] = *reinterpret_cast<const v8bf*>(&As[wm + 16 + ln][8 * half]);
        a1.h[1] = *reinterpret_cast<const v8bf*>(&As[wm + 16 + ln][16 + 8 * half]);
        b0.h[0] = *reinterpret_cast<const v8bf*>(&Bs[wn + ln][8 * half]);
        b0.h[1] = *reinterpret_cast<const v8bf*>(&Bs[wn + ln][16 + 8 * half]);
        b1.h[0] = *reinterpret_cast<const v8bf*>(&Bs[wn + 16 + ln][8 * half]);
        b1.h[1] = *reinterpret_cast<const v8bf*>(&Bs[wn + 16 + ln][16 + 8 * half]);

        c[0][0] = __builtin_amdgcn_wmma_f32_16x16x32_bf16(false, a0.v, false, b0.v, (short)0, c[0][0], false, false);
        c[0][1] = __builtin_amdgcn_wmma_f32_16x16x32_bf16(false, a0.v, false, b1.v, (short)0, c[0][1], false, false);
        c[1][0] = __builtin_amdgcn_wmma_f32_16x16x32_bf16(false, a1.v, false, b0.v, (short)0, c[1][0], false, false);
        c[1][1] = __builtin_amdgcn_wmma_f32_16x16x32_bf16(false, a1.v, false, b1.v, (short)0, c[1][1], false, false);
        __syncthreads();
    }

    #pragma unroll
    for (int mt = 0; mt < 2; ++mt) {
        #pragma unroll
        for (int nt = 0; nt < 2; ++nt) {
            const int gn = n0 + wn + nt * 16 + ln;
            const float bval = bia[gn];
            #pragma unroll
            for (int r = 0; r < 8; ++r) {
                const int gm = m0 + wm + mt * 16 + r + 8 * half;
                const float v = c[mt][nt][r] + bval;
                if (TRANSPOSED) outp[(size_t)gn * SEQ    + gm] = (bf16_t)v;
                else            outp[(size_t)gm * DMODEL + gn] = (bf16_t)v;
            }
        }
    }
}

// ---------------------------------------------------------------------------
// Kernel 2: scores + softmax, writing attention = softmax(QK^T/32)^T.
// Computes T[k][q] = S[q][k] tiles directly (A = K rows, B = Q rows) so the
// fp32 attention write att[h][k][q0+lane] is coalesced. Each wave keeps a
// 16(q) x 256(k) strip in 16 accumulator fragments; block covers k = 0..2047.
// ---------------------------------------------------------------------------
__global__ __launch_bounds__(256) void attn_softmax_kernel(
    const bf16_t* __restrict__ Qb, const bf16_t* __restrict__ Kb,
    float* __restrict__ att)
{
    const int t    = threadIdx.x;
    const int lane = t & 31;
    const int wave = t >> 5;      // 0..7
    const int half = lane >> 4;
    const int ln   = lane & 15;

    const int q0 = blockIdx.x * 16;
    const int h  = blockIdx.y;
    const int cb = h * HD;        // head column base in Qb/Kb

    __shared__ float redmax[8][16];
    __shared__ float redsum[8][16];
    __shared__ float colmax[16];
    __shared__ float colsum[16];

    // B fragments: Q rows q0+ln, d = 0..63 (2 frags of depth 32), load once
    const bf16_t* qrow = Qb + (size_t)(q0 + ln) * DMODEL + cb;
    V16U bq0, bq1;
    bq0.h[0] = *(const v8bf*)(qrow + 8 * half);
    bq0.h[1] = *(const v8bf*)(qrow + 16 + 8 * half);
    bq1.h[0] = *(const v8bf*)(qrow + 32 + 8 * half);
    bq1.h[1] = *(const v8bf*)(qrow + 48 + 8 * half);

    v8f c[16];
    const int kw = wave * 256;
    #pragma unroll
    for (int kt = 0; kt < 16; ++kt) {
        const bf16_t* krow = Kb + (size_t)(kw + kt * 16 + ln) * DMODEL + cb;
        V16U a0, a1;
        a0.h[0] = *(const v8bf*)(krow + 8 * half);
        a0.h[1] = *(const v8bf*)(krow + 16 + 8 * half);
        a1.h[0] = *(const v8bf*)(krow + 32 + 8 * half);
        a1.h[1] = *(const v8bf*)(krow + 48 + 8 * half);
        v8f acc = {};
        acc = __builtin_amdgcn_wmma_f32_16x16x32_bf16(false, a0.v, false, bq0.v, (short)0, acc, false, false);
        acc = __builtin_amdgcn_wmma_f32_16x16x32_bf16(false, a1.v, false, bq1.v, (short)0, acc, false, false);
        #pragma unroll
        for (int r = 0; r < 8; ++r) acc[r] *= 0.03125f;   // 1/sqrt(1024)
        c[kt] = acc;
    }

    // softmax over k (tile rows) per column q = q0 + ln
    float m = -3.4e38f;
    #pragma unroll
    for (int kt = 0; kt < 16; ++kt)
        #pragma unroll
        for (int r = 0; r < 8; ++r) m = fmaxf(m, c[kt][r]);
    m = fmaxf(m, __shfl_xor(m, 16, 32));       // combine row halves
    if (lane < 16) redmax[wave][ln] = m;
    __syncthreads();
    if (t < 16) {
        float mm = redmax[0][t];
        for (int w = 1; w < 8; ++w) mm = fmaxf(mm, redmax[w][t]);
        colmax[t] = mm;
    }
    __syncthreads();
    const float cm = colmax[ln];

    float s = 0.0f;
    #pragma unroll
    for (int kt = 0; kt < 16; ++kt)
        #pragma unroll
        for (int r = 0; r < 8; ++r) {
            float e = __expf(c[kt][r] - cm);
            c[kt][r] = e;
            s += e;
        }
    s += __shfl_xor(s, 16, 32);
    if (lane < 16) redsum[wave][ln] = s;
    __syncthreads();
    if (t < 16) {
        float ss = 0.0f;
        for (int w = 0; w < 8; ++w) ss += redsum[w][t];
        colsum[t] = ss;
    }
    __syncthreads();
    const float inv = 1.0f / colsum[ln];

    // att[h][k][q] = P[q][k]; 16 consecutive lanes -> 64B coalesced stores
    float* base = att + (size_t)h * SEQ * SEQ + q0 + ln;
    #pragma unroll
    for (int kt = 0; kt < 16; ++kt) {
        #pragma unroll
        for (int r = 0; r < 8; ++r) {
            const int k = kw + kt * 16 + r + 8 * half;
            base[(size_t)k * SEQ] = c[kt][r] * inv;
        }
    }
}

// ---------------------------------------------------------------------------
// Kernel 3: av = attention_matrix @ V  (per head), A read fp32 -> bf16.
// Block = 32(i) x 64(d) per head, 8 waves each owning one 16x16 tile.
// Lane pairs (L, L+16) read the same att row at complementary k offsets, so
// each 128B cacheline of the attention matrix is consumed exactly once.
// ---------------------------------------------------------------------------
__global__ __launch_bounds__(256) void av_gemm_kernel(
    const float* __restrict__ att, const bf16_t* __restrict__ Vt,
    float* __restrict__ out)
{
    const int t    = threadIdx.x;
    const int lane = t & 31;
    const int wave = t >> 5;
    const int half = lane >> 4;
    const int ln   = lane & 15;

    const int h  = blockIdx.y;
    const int i0 = blockIdx.x * 32;

    const int irow = i0 + (wave & 1) * 16;     // i tile base
    const int dcol = (wave >> 1) * 16;         // d tile base (0..48)

    const float*  arow = att + ((size_t)h * SEQ + irow + ln) * SEQ;
    const bf16_t* brow = Vt  + (size_t)(h * HD + dcol + ln) * SEQ;

    v8f c = {};
    for (int kb = 0; kb < SEQ; kb += 32) {
        v4f f0 = *(const v4f*)(arow + kb + 8 * half);
        v4f f1 = *(const v4f*)(arow + kb + 8 * half + 4);
        v4f f2 = *(const v4f*)(arow + kb + 16 + 8 * half);
        v4f f3 = *(const v4f*)(arow + kb + 16 + 8 * half + 4);
        V16U a;
        #pragma unroll
        for (int i = 0; i < 4; ++i) {
            a.h[0][i]   = (bf16_t)f0[i]; a.h[0][i+4] = (bf16_t)f1[i];
            a.h[1][i]   = (bf16_t)f2[i]; a.h[1][i+4] = (bf16_t)f3[i];
        }
        V16U b;
        b.h[0] = *(const v8bf*)(brow + kb + 8 * half);
        b.h[1] = *(const v8bf*)(brow + kb + 16 + 8 * half);
        c = __builtin_amdgcn_wmma_f32_16x16x32_bf16(false, a.v, false, b.v, (short)0, c, false, false);
    }

    // output flat layout == av flat: h*SEQ*HD + i*HD + d
    #pragma unroll
    for (int r = 0; r < 8; ++r) {
        const int i = irow + r + 8 * half;
        const int d = dcol + ln;
        out[(size_t)h * SEQ * HD + (size_t)i * HD + d] = c[r];
    }
}

// ---------------------------------------------------------------------------
extern "C" void kernel_launch(void* const* d_in, const int* in_sizes, int n_in,
                              void* d_out, int out_size, void* d_ws, size_t ws_size,
                              hipStream_t stream)
{
    (void)in_sizes; (void)n_in; (void)out_size; (void)ws_size;

    const float* x  = (const float*)d_in[0];
    const float* Wq = (const float*)d_in[1];
    const float* bq = (const float*)d_in[2];
    const float* Wk = (const float*)d_in[3];
    const float* bk = (const float*)d_in[4];
    const float* Wv = (const float*)d_in[5];
    const float* bv = (const float*)d_in[6];

    // workspace: bf16 Q, K (row-major) and V transposed — 12 MB total
    bf16_t* Qb = (bf16_t*)d_ws;
    bf16_t* Kb = Qb + (size_t)SEQ * DMODEL;
    bf16_t* Vt = Kb + (size_t)SEQ * DMODEL;

    float* att = (float*)d_out;                       // [16][2048][2048] fp32
    float* out = att + (size_t)NH * SEQ * SEQ;        // [2048][1024] fp32 (flat av)

    dim3 g1(DMODEL / 64, SEQ / 128);
    hipLaunchKernelGGL((proj_gemm_kernel<false>), g1, dim3(256), 0, stream, x, Wq, bq, Qb);
    hipLaunchKernelGGL((proj_gemm_kernel<false>), g1, dim3(256), 0, stream, x, Wk, bk, Kb);
    hipLaunchKernelGGL((proj_gemm_kernel<true>),  g1, dim3(256), 0, stream, x, Wv, bv, Vt);

    dim3 g2(SEQ / 16, NH);
    hipLaunchKernelGGL(attn_softmax_kernel, g2, dim3(256), 0, stream, Qb, Kb, att);

    dim3 g3(SEQ / 32, NH);
    hipLaunchKernelGGL(av_gemm_kernel, g3, dim3(256), 0, stream, att, Vt, out);
}
